// PointnetSAModuleVotes_group_enc_v2_42125039239558
// MI455X (gfx1250) — matrix operand order repaired and checked by reading, not compiled, not run on hardware
//
#include <hip/hip_runtime.h>
#include <hip/hip_bf16.h>

typedef __attribute__((ext_vector_type(16))) _Float16 v16h;
typedef __attribute__((ext_vector_type(8)))  float    v8f;

#define NN 16384
#define BB 4
#define NP 1024
#define NS 64
#define R2 0.09f
#define MTOT (BB*NP*NS)   // 262144 rows into the MLP

// ---------------------------------------------------------------------------
// 1) Farthest point sampling: one workgroup per batch, xyz cached in registers
// ---------------------------------------------------------------------------
__global__ __launch_bounds__(1024) void fps_kernel(const float* __restrict__ xyz,
                                                   float* __restrict__ new_xyz,
                                                   int* __restrict__ inds) {
  const int b = blockIdx.x;
  const int t = threadIdx.x;
  __shared__ float sval[1024];
  __shared__ int   sidx[1024];
  __shared__ float sp[3];
  const float* bx = xyz + (size_t)b * NN * 3;
  float px[16], py[16], pz[16], mind[16];
#pragma unroll
  for (int i = 0; i < 16; ++i) {
    int pid = t + i * 1024;
    px[i] = bx[pid*3+0]; py[i] = bx[pid*3+1]; pz[i] = bx[pid*3+2];
    mind[i] = 1e10f;
  }
  int last = 0;
  for (int it = 0; it < NP; ++it) {
    if (t == 0) {
      inds[b*NP + it] = last;
      float qx = bx[last*3+0], qy = bx[last*3+1], qz = bx[last*3+2];
      new_xyz[(b*NP+it)*3+0] = qx;
      new_xyz[(b*NP+it)*3+1] = qy;
      new_xyz[(b*NP+it)*3+2] = qz;
      sp[0] = qx; sp[1] = qy; sp[2] = qz;
    }
    __syncthreads();
    float qx = sp[0], qy = sp[1], qz = sp[2];
    float bv = -1.f; int bi = 0;
#pragma unroll
    for (int i = 0; i < 16; ++i) {
      float dx = px[i]-qx, dy = py[i]-qy, dz = pz[i]-qz;
      float d = dx*dx + dy*dy + dz*dz;
      mind[i] = fminf(mind[i], d);
      if (mind[i] > bv) { bv = mind[i]; bi = t + i*1024; }
    }
    sval[t] = bv; sidx[t] = bi;
    __syncthreads();
    for (int s = 512; s > 0; s >>= 1) {
      if (t < s) {
        float v2 = sval[t+s]; int i2 = sidx[t+s];
        if (v2 > sval[t] || (v2 == sval[t] && i2 < sidx[t])) { sval[t] = v2; sidx[t] = i2; }
      }
      __syncthreads();
    }
    last = sidx[0];
    __syncthreads();
  }
}

// ---------------------------------------------------------------------------
// 2) Ball query + gather: one wave32 per query point; ordered compaction via
//    wave32 ballot; writes the f16 MLP input rows (xyz residual ++ features)
// ---------------------------------------------------------------------------
__global__ __launch_bounds__(256) void ballq_kernel(const float* __restrict__ xyz,
                                                    const float* __restrict__ feat,
                                                    const float* __restrict__ new_xyz,
                                                    _Float16* __restrict__ X) {
  const int lane = threadIdx.x & 31;
  const int wave = threadIdx.x >> 5;
  const int q = blockIdx.x * 8 + wave;          // 0..4095 == b*1024 + p
  const int b = q >> 10;
  const float* bx = xyz  + (size_t)b * NN * 3;
  const float* bf = feat + (size_t)b * 3 * NN;
  float nx = new_xyz[q*3+0], ny = new_xyz[q*3+1], nz = new_xyz[q*3+2];
  _Float16* row0 = X + (size_t)q * NS * 6;
  int found = 0, firstIdx = -1;
  for (int c = 0; c < NN/32 && found < NS; ++c) {
    int pid = c*32 + lane;
    float dx = bx[pid*3+0]-nx, dy = bx[pid*3+1]-ny, dz = bx[pid*3+2]-nz;
    float d = dx*dx + dy*dy + dz*dz;
    bool pred = d < R2;
    unsigned mask = (unsigned)__builtin_amdgcn_ballot_w32(pred);
    if (firstIdx < 0 && mask) firstIdx = c*32 + __builtin_ctz(mask);
    if (pred) {
      int slot = found + __popc(mask & ((1u << lane) - 1u));
      if (slot < NS) {
        _Float16* r = row0 + slot*6;
        r[0] = (_Float16)dx; r[1] = (_Float16)dy; r[2] = (_Float16)dz;
        r[3] = (_Float16)bf[0*NN + pid];
        r[4] = (_Float16)bf[1*NN + pid];
        r[5] = (_Float16)bf[2*NN + pid];
      }
    }
    found += __popc(mask);
  }
  if (found < NS) {                // pad with first candidate (reference semantics)
    int fi = (firstIdx < 0) ? 0 : firstIdx;
    float dx = bx[fi*3+0]-nx, dy = bx[fi*3+1]-ny, dz = bx[fi*3+2]-nz;
    float f0 = bf[0*NN+fi], f1 = bf[1*NN+fi], f2 = bf[2*NN+fi];
    for (int s = found + lane; s < NS; s += 32) {
      _Float16* r = row0 + s*6;
      r[0] = (_Float16)dx; r[1] = (_Float16)dy; r[2] = (_Float16)dz;
      r[3] = (_Float16)f0; r[4] = (_Float16)f1; r[5] = (_Float16)f2;
    }
  }
}

// ---------------------------------------------------------------------------
// WMMA fragment handling (CDNA5 ISA §7.12.2, wave32)
// ---------------------------------------------------------------------------
// A 16x32 f16 (MxK): lane L -> M = L&15;
// halves [0..7] = K {0..7}+khalf, halves [8..15] = K {16..23}+khalf,
// khalf = 8 for lanes 16..31. Template KLIM so OOB halves fold to constants
// and in-bounds runs coalesce into contiguous loads.
template <int KLIM>
__device__ inline v16h load_a_frag(const _Float16* __restrict__ Hin, int m0, int lane,
                                   int kofs) {
  v16h a;
  const int m = m0 + (lane & 15);
  const int khalf = (lane & 16) ? 8 : 0;
  const _Float16* rp = Hin + (size_t)m * KLIM;
#pragma unroll
  for (int hi = 0; hi < 16; ++hi) {
    int v = hi >> 1;
    int k = ((v & 3) << 1) + (hi & 1) + ((v >> 2) << 4) + khalf + kofs;
    a[hi] = (k < KLIM) ? rp[k] : (_Float16)0.f;
  }
  return a;
}

// Stage W (NOUT x KLIM, f32) into LDS pre-packed in B-fragment order:
// WtF[(nt*KSTEPS + ks)*32 + lane] is the v16h fragment for that (nt,ks).
// B 32x16 layout: lane L -> N = nt*16 + (L&15); K = (hi&7) + (hi&8?16:0) + (L&16?8:0) + ks*32.
template <int KLIM, int NOUT, int KSTEPS>
__device__ inline void fill_wt(v16h* WtF, const float* __restrict__ W, int tid) {
  _Float16* Wth = (_Float16*)WtF;
  const int total = (NOUT / 16) * KSTEPS * 32 * 16;
  for (int i = tid; i < total; i += 256) {
    int hi = i & 15;
    int L  = (i >> 4) & 31;
    int f  = i >> 9;
    int ks = f % KSTEPS;
    int nt = f / KSTEPS;
    int n  = nt * 16 + (L & 15);
    int k  = (hi & 7) + ((hi & 8) ? 16 : 0) + ((L & 16) ? 8 : 0) + ks * 32;
    Wth[i] = (k < KLIM) ? (_Float16)W[n * KLIM + k] : (_Float16)0.f;
  }
}

// ---------------------------------------------------------------------------
// 3) MLP layer: Hout = relu(Hin @ W^T + b), f16 store, f32 accumulate.
//    One wave per 16-row M-tile; fully unrolled via templates.
// ---------------------------------------------------------------------------
template <int KLIM, int NOUT, int KSTEPS>
__global__ __launch_bounds__(256) void mlp_layer_kernel(const _Float16* __restrict__ Hin,
                                                        const float* __restrict__ W,
                                                        const float* __restrict__ bias,
                                                        _Float16* __restrict__ Hout) {
  __shared__ v16h WtF[(NOUT / 16) * KSTEPS * 32];
  fill_wt<KLIM, NOUT, KSTEPS>(WtF, W, threadIdx.x);
  __syncthreads();
  const int lane = threadIdx.x & 31, wave = threadIdx.x >> 5;
  const int m0 = (blockIdx.x * 8 + wave) * 16;
  __builtin_prefetch(Hin + (size_t)m0 * KLIM, 0, 1);   // global_prefetch_b8
  v16h a[KSTEPS];
#pragma unroll
  for (int ks = 0; ks < KSTEPS; ++ks)
    a[ks] = load_a_frag<KLIM>(Hin, m0, lane, ks * 32);
#pragma unroll
  for (int nt = 0; nt < NOUT / 16; ++nt) {
    float bv = bias[nt * 16 + (lane & 15)];
    v8f acc = { bv, bv, bv, bv, bv, bv, bv, bv };
#pragma unroll
    for (int ks = 0; ks < KSTEPS; ++ks) {
      v16h bf = WtF[(nt * KSTEPS + ks) * 32 + lane];
      acc = __builtin_amdgcn_wmma_f32_16x16x32_f16(false, a[ks], false, bf,
                                                   (short)0, acc, false, false);
    }
    const int n  = nt * 16 + (lane & 15);
    const int mo = (lane & 16) ? 8 : 0;
#pragma unroll
    for (int v = 0; v < 8; ++v)
      Hout[(size_t)(m0 + v + mo) * NOUT + n] = (_Float16)fmaxf(acc[v], 0.f);
  }
}

// ---------------------------------------------------------------------------
// 4) Layer 2 (64->128) fused with max-pool over the 64 samples of each point.
//    One wave per point; mt loop kept rolled to bound register pressure.
// ---------------------------------------------------------------------------
__global__ __launch_bounds__(256) void mlp_layer2_pool_kernel(const _Float16* __restrict__ H2,
                                                              const float* __restrict__ W2,
                                                              const float* __restrict__ b2,
                                                              float* __restrict__ pooled) {
  __shared__ v16h WtF[8 * 2 * 32];
  fill_wt<64, 128, 2>(WtF, W2, threadIdx.x);
  __syncthreads();
  const int lane = threadIdx.x & 31, wave = threadIdx.x >> 5;
  const int q = blockIdx.x * 8 + wave;       // point id: b*1024 + p
  const int b = q >> 10, p = q & 1023;
  v8f maxv[8];
#pragma unroll
  for (int nt = 0; nt < 8; ++nt) {
    v8f z = {0.f,0.f,0.f,0.f,0.f,0.f,0.f,0.f};   // relu output >= 0
    maxv[nt] = z;
  }
  for (int mt = 0; mt < 4; ++mt) {
    const int m0 = q * 64 + mt * 16;
    v16h a0 = load_a_frag<64>(H2, m0, lane, 0);
    v16h a1 = load_a_frag<64>(H2, m0, lane, 32);
#pragma unroll
    for (int nt = 0; nt < 8; ++nt) {
      float bv = b2[nt * 16 + (lane & 15)];
      v8f acc = { bv, bv, bv, bv, bv, bv, bv, bv };
      v16h bf0 = WtF[(nt * 2 + 0) * 32 + lane];
      acc = __builtin_amdgcn_wmma_f32_16x16x32_f16(false, a0, false, bf0,
                                                   (short)0, acc, false, false);
      v16h bf1 = WtF[(nt * 2 + 1) * 32 + lane];
      acc = __builtin_amdgcn_wmma_f32_16x16x32_f16(false, a1, false, bf1,
                                                   (short)0, acc, false, false);
#pragma unroll
      for (int v = 0; v < 8; ++v)
        maxv[nt][v] = fmaxf(maxv[nt][v], fmaxf(acc[v], 0.f));
    }
  }
#pragma unroll
  for (int nt = 0; nt < 8; ++nt) {
    float mv = maxv[nt][0];
#pragma unroll
    for (int v = 1; v < 8; ++v) mv = fmaxf(mv, maxv[nt][v]);
    mv = fmaxf(mv, __shfl_xor(mv, 16, 32));    // combine M and M+8 halves
    if (lane < 16) {
      int n = (nt << 4) + lane;
      pooled[((size_t)b * 128 + n) * NP + p] = mv;
    }
  }
}

// ---------------------------------------------------------------------------
// 5) Encoder fuse: per (g,b) block. Uses softmax identity:
//    sum_k a_k (x - cw_k) = x - sum_k a_k cw_k
// ---------------------------------------------------------------------------
__global__ __launch_bounds__(256) void enc_fuse_kernel(const float* __restrict__ pooled,
                                                       const float* __restrict__ codewords,
                                                       const float* __restrict__ scale,
                                                       const float* __restrict__ fc_w,
                                                       const float* __restrict__ fc_b,
                                                       float* __restrict__ fout) {
  const int g = blockIdx.x & 3;
  const int b = blockIdx.x >> 2;
  const int tid = threadIdx.x;
  __shared__ float cw[32*32];
  __shared__ float sc[32];
  __shared__ float e[32];
  __shared__ float gamma[32];
  for (int i = tid; i < 32*32; i += 256) cw[i] = codewords[g*1024 + i];
  if (tid < 32) { sc[tid] = scale[g*32 + tid]; e[tid] = 0.f; }
  __syncthreads();
  float le[32];
#pragma unroll
  for (int d = 0; d < 32; ++d) le[d] = 0.f;
  for (int p = tid; p < NP; p += 256) {
    float x[32];
#pragma unroll
    for (int d = 0; d < 32; ++d)
      x[d] = pooled[((size_t)b*128 + d*4 + g) * NP + p];
    float a[32]; float mmax = -1e30f;
#pragma unroll
    for (int k = 0; k < 32; ++k) {
      float dist = 0.f;
#pragma unroll
      for (int d = 0; d < 32; ++d) { float r = x[d] - cw[k*32+d]; dist += r*r; }
      float t = -sc[k] * dist;
      a[k] = t;
      mmax = fmaxf(mmax, t);
    }
    float ssum = 0.f;
#pragma unroll
    for (int k = 0; k < 32; ++k) { a[k] = __expf(a[k] - mmax); ssum += a[k]; }
    float inv = 1.f / ssum;
#pragma unroll
    for (int d = 0; d < 32; ++d) {
      float t = 0.f;
#pragma unroll
      for (int k = 0; k < 32; ++k) t += a[k] * cw[k*32+d];
      le[d] += x[d] - t * inv;
    }
  }
#pragma unroll
  for (int d = 0; d < 32; ++d) atomicAdd(&e[d], le[d]);
  __syncthreads();
  if (tid < 32) {
    float acc = fc_b[g*32 + tid];
#pragma unroll
    for (int j = 0; j < 32; ++j)
      acc += fc_w[((size_t)g*32 + tid)*32 + j] * (e[j] * (1.f/32.f)); // mean over K
    gamma[tid] = 1.f / (1.f + __expf(-acc));
  }
  __syncthreads();
  for (int i = tid; i < 32*NP; i += 256) {
    int d = i >> 10, p = i & 1023;
    size_t off = ((size_t)b*128 + d*4 + g) * NP + p;
    fout[off] = pooled[off] * gamma[d];
  }
}

// ---------------------------------------------------------------------------
// Launch: d_out = [new_xyz (B,1024,3) f32 | features_out (B,128,1024) f32 |
//                  fps_inds (B,1024) i32]
// Workspace (~72.4 MB): X f16 | H1 f16 | H2 f16 | pooled f32
// ---------------------------------------------------------------------------
extern "C" void kernel_launch(void* const* d_in, const int* in_sizes, int n_in,
                              void* d_out, int out_size, void* d_ws, size_t ws_size,
                              hipStream_t stream) {
  (void)in_sizes; (void)n_in; (void)out_size; (void)ws_size;
  const float* xyz       = (const float*)d_in[0];
  const float* feat      = (const float*)d_in[1];
  const float* w0        = (const float*)d_in[2];
  const float* b0        = (const float*)d_in[3];
  const float* w1        = (const float*)d_in[4];
  const float* b1        = (const float*)d_in[5];
  const float* w2        = (const float*)d_in[6];
  const float* b2        = (const float*)d_in[7];
  const float* codewords = (const float*)d_in[8];
  const float* scale     = (const float*)d_in[9];
  const float* fc_w      = (const float*)d_in[10];
  const float* fc_b      = (const float*)d_in[11];

  float* out_newxyz = (float*)d_out;                  // 12288 floats
  float* out_feat   = out_newxyz + BB*NP*3;           // 524288 floats
  int*   out_inds   = (int*)(out_feat + BB*128*NP);   // 4096 ints

  char* ws = (char*)d_ws;
  _Float16* X  = (_Float16*)ws;
  size_t offH1 = (size_t)MTOT * 6  * sizeof(_Float16);
  _Float16* H1 = (_Float16*)(ws + offH1);
  size_t offH2 = offH1 + (size_t)MTOT * 64 * sizeof(_Float16);
  _Float16* H2 = (_Float16*)(ws + offH2);
  size_t offP  = offH2 + (size_t)MTOT * 64 * sizeof(_Float16);
  float* pooled = (float*)(ws + offP);

  fps_kernel<<<BB, 1024, 0, stream>>>(xyz, out_newxyz, out_inds);
  ballq_kernel<<<512, 256, 0, stream>>>(xyz, feat, out_newxyz, X);
  mlp_layer_kernel<6, 64, 1><<<2048, 256, 0, stream>>>(X,  w0, b0, H1);
  mlp_layer_kernel<64, 64, 2><<<2048, 256, 0, stream>>>(H1, w1, b1, H2);
  mlp_layer2_pool_kernel<<<512, 256, 0, stream>>>(H2, w2, b2, pooled);
  enc_fuse_kernel<<<16, 256, 0, stream>>>(pooled, codewords, scale, fc_w, fc_b, out_feat);
}
